// GridTorch_81862076661760
// MI455X (gfx1250) — matrix operand order repaired and checked by reading, not compiled, not run on hardware
//
#include <hip/hip_runtime.h>
#include <hip/hip_bf16.h>

typedef __attribute__((ext_vector_type(16))) _Float16 v16h;
typedef __attribute__((ext_vector_type(8)))  float    v8f;

#define T_STEPS 1000
#define BB      128
#define NH      256
#define NIC     268
#define NPC     256
#define NHD     12
#define NW      268              // NPC + NHD
#define ROWS    (T_STEPS * BB)   // 128000
#define MPW     8                // M-tiles per wave (B frags reused)

// ---------------------------------------------------------------------------
// Kernel 1: fused init-embed + SNU recurrence scan.
// One thread per (b, h). Carry (y, s) lives in registers across all T steps.
// ---------------------------------------------------------------------------
__global__ void __launch_bounds__(256)
snu_scan_kernel(const float* __restrict__ x,     // [T, B, 3]
                const float* __restrict__ ic,    // [B, 268]
                const float* __restrict__ iw,    // [3, 256]
                const float* __restrict__ lv,    // [256]
                const float* __restrict__ bv,    // [256]
                const float* __restrict__ sew,   // [268, 256]
                const float* __restrict__ seb,   // [256]
                const float* __restrict__ cew,   // [268, 256]
                const float* __restrict__ ceb,   // [256]
                float* __restrict__ out_y_bn,    // [T, B, 256]
                float* __restrict__ out_y_rnn,   // [T, B, 256]
                float* __restrict__ out_s)       // [T, B, 256]
{
    const int b = blockIdx.x;    // 0..127
    const int h = threadIdx.x;   // 0..255

    const float* icb = ic + (size_t)b * NIC;
    float hacc = seb[h];
    float cacc = ceb[h];
    #pragma unroll 4
    for (int j = 0; j < NIC; ++j) {
        const float v = icb[j];
        hacc = fmaf(v, sew[(size_t)j * NH + h], hacc);
        cacc = fmaf(v, cew[(size_t)j * NH + h], cacc);
    }
    float y = 1.0f / (1.0f + __expf(-hacc));
    float c = 1.0f / (1.0f + __expf(-cacc));

    const float w0 = iw[0 * NH + h];
    const float w1 = iw[1 * NH + h];
    const float w2 = iw[2 * NH + h];
    const float lh = lv[h];
    const float bh = bv[h];

    size_t idx = (size_t)b * NH + h;
    const float* xb = x + (size_t)b * 3;

    for (int t = 0; t < T_STEPS; ++t) {
        const float* xt = xb + (size_t)t * BB * 3;  // block-uniform -> scalar loads
        float z = fmaf(xt[0], w0, fmaf(xt[1], w1, xt[2] * w2));
        float s = z + y + lh * c * (1.0f - y);
        s = fmaxf(s, 0.0f);
        const float ynew = (s + bh > 0.0f) ? 1.0f : 0.0f;
        out_s[idx]     = s;
        out_y_bn[idx]  = ynew;
        out_y_rnn[idx] = ynew;
        y = ynew;
        c = s;
        idx += (size_t)BB * NH;
    }
}

// ---------------------------------------------------------------------------
// Kernel 2: pack [pc_w | hd_w] into f16, transposed to n-major [268][256].
// ---------------------------------------------------------------------------
__global__ void __launch_bounds__(256)
pack_w16_kernel(const float* __restrict__ pc_w,  // [256, 256]
                const float* __restrict__ hd_w,  // [256, 12]
                _Float16* __restrict__ w16)      // [268][256], n-major
{
    const int idx = blockIdx.x * 256 + threadIdx.x;
    if (idx >= NW * NH) return;
    const int n = idx / NH;   // output column 0..267
    const int k = idx % NH;   // input row 0..255
    const float v = (n < NPC) ? pc_w[(size_t)k * NPC + n]
                              : hd_w[(size_t)k * NHD + (n - NPC)];
    w16[idx] = (_Float16)v;
}

// ---------------------------------------------------------------------------
// SSA-friendly staged A fragment: 16 raw f32 (one WMMA k-step for this lane),
// kept strictly by value so SROA keeps everything in registers.
// ---------------------------------------------------------------------------
struct A4 { float4 a, b, c, d; };

__device__ __forceinline__ A4 load_a(const float4* __restrict__ Y4,
                                     size_t base4, int mm, int kk) {
    // float4 offset: row*64 + kk*8 + hi*2 (hi folded into base4);
    // halves 0..7 -> {o, o+1}, halves 8..15 -> {o+4, o+5}
    const size_t o = base4 + (size_t)mm * (16 * 64) + (size_t)kk * 8;
    A4 r;
    r.a = Y4[o];     r.b = Y4[o + 1];
    r.c = Y4[o + 4]; r.d = Y4[o + 5];
    return r;
}

__device__ __forceinline__ v16h cvt_a(A4 r) {
    v16h af;
    af[0]  = (_Float16)r.a.x; af[1]  = (_Float16)r.a.y;
    af[2]  = (_Float16)r.a.z; af[3]  = (_Float16)r.a.w;
    af[4]  = (_Float16)r.b.x; af[5]  = (_Float16)r.b.y;
    af[6]  = (_Float16)r.b.z; af[7]  = (_Float16)r.b.w;
    af[8]  = (_Float16)r.c.x; af[9]  = (_Float16)r.c.y;
    af[10] = (_Float16)r.c.z; af[11] = (_Float16)r.c.w;
    af[12] = (_Float16)r.d.x; af[13] = (_Float16)r.d.y;
    af[14] = (_Float16)r.d.z; af[15] = (_Float16)r.d.w;
    return af;
}

// ---------------------------------------------------------------------------
// Kernel 3: batched readout GEMM with WMMA.
//   [ROWS, 256] (spikes, f32 -> f16) x [256, 268] (f16) -> pc & hd (f32)
// One wave owns one N-tile and MPW consecutive M-tiles. B fragments loaded
// once into 64 VGPRs. A is software-pipelined one k-step ahead via a pure
// SSA cur/nxt rotation (no arrays -> no scratch/LDS demotion), and the
// prefetch crosses m-tile boundaries in the kk==7 slot.
// ---------------------------------------------------------------------------
__global__ void __launch_bounds__(256)
spike_readout_gemm(const float* __restrict__ yv,    // [ROWS, 256]
                   const _Float16* __restrict__ w16,// [268][256] n-major
                   const float* __restrict__ pc_b,  // [256]
                   const float* __restrict__ hd_b,  // [12]
                   float* __restrict__ out_pc,      // [ROWS, 256]
                   float* __restrict__ out_hd)      // [ROWS, 12]
{
    const int lane   = threadIdx.x & 31;
    const int wave   = threadIdx.x >> 5;              // 0..7
    const int ntile  = blockIdx.x;                    // 0..16
    const int mtile0 = (blockIdx.y * 8 + wave) * MPW; // base of MPW tiles
    const int n0     = ntile * 16;
    const int mloc   = lane & 15;
    const int hi     = lane >> 4;                     // 0 or 1

    // ---- B fragments: element i<8 holds K=hi*8+i, i>=8 holds K=16+hi*8+(i-8),
    //      N = lane & 15. Loaded once. ---------
    const int  n    = n0 + mloc;
    const bool nvld = (n < NW);
    const int  ncl  = nvld ? n : (NW - 1);            // clamp, keep EXEC uniform
    const _Float16 zh = (_Float16)0.0f;

    v16h bfrag[8];
    #pragma unroll
    for (int kk = 0; kk < 8; ++kk) {
        const _Float16* wp = w16 + (size_t)ncl * NH + kk * 32 + hi * 8;
        v16h bf;
        #pragma unroll
        for (int i = 0; i < 16; ++i)
            bf[i] = nvld ? wp[i + (i & 8)] : zh;
        bfrag[kk] = bf;
    }

    const float bias = (ntile < 16) ? pc_b[n] : (nvld ? hd_b[n - NPC] : 0.0f);

    const float4* Y4 = (const float4*)yv;
    const size_t base4 = (size_t)(mtile0 * 16 + mloc) * 64 + hi * 2;

    A4 cur = load_a(Y4, base4, 0, 0);

    for (int mm = 0; mm < MPW; ++mm) {
        v8f acc = {};
        #pragma unroll
        for (int kk = 0; kk < 8; ++kk) {
            // prefetch the next k-step (crossing into the next m-tile at kk==7)
            A4 nxt;
            if (kk < 7)            nxt = load_a(Y4, base4, mm, kk + 1);
            else if (mm + 1 < MPW) nxt = load_a(Y4, base4, mm + 1, 0);
            else                   nxt = cur;  // drained

            const v16h af = cvt_a(cur);
            acc = __builtin_amdgcn_wmma_f32_16x16x32_f16(
                      false, af, false, bfrag[kk], (short)0, acc, false, false);
            cur = nxt;
        }

        // flush tile: element i -> (M = row0 + i + hi*8, N = n)
        const int row0 = (mtile0 + mm) * 16;
        if (ntile < 16) {
            #pragma unroll
            for (int i = 0; i < 8; ++i)
                out_pc[(size_t)(row0 + i + hi * 8) * NPC + n] = acc[i] + bias;
        } else if (nvld) {
            const int nh = n - NPC;                   // 0..11
            #pragma unroll
            for (int i = 0; i < 8; ++i)
                out_hd[(size_t)(row0 + i + hi * 8) * NHD + nh] = acc[i] + bias;
        }
    }
}

// ---------------------------------------------------------------------------
extern "C" void kernel_launch(void* const* d_in, const int* in_sizes, int n_in,
                              void* d_out, int out_size, void* d_ws, size_t ws_size,
                              hipStream_t stream) {
    (void)in_sizes; (void)n_in; (void)out_size; (void)ws_size;

    const float* x    = (const float*)d_in[0];   // [1000,128,3]
    const float* ic   = (const float*)d_in[1];   // [128,268]
    const float* iw   = (const float*)d_in[2];   // [3,256]
    const float* lv   = (const float*)d_in[3];   // [256]
    const float* bv   = (const float*)d_in[4];   // [256]
    const float* sew  = (const float*)d_in[5];   // [268,256]
    const float* seb  = (const float*)d_in[6];   // [256]
    const float* cew  = (const float*)d_in[7];   // [268,256]
    const float* ceb  = (const float*)d_in[8];   // [256]
    const float* pc_w = (const float*)d_in[9];   // [256,256]
    const float* pc_b = (const float*)d_in[10];  // [256]
    const float* hd_w = (const float*)d_in[11];  // [256,12]
    const float* hd_b = (const float*)d_in[12];  // [12]

    float* out = (float*)d_out;
    float* out_hd   = out;                              // [T,B,12]
    float* out_pc   = out + (size_t)ROWS * NHD;         // [T,B,256]
    float* out_y_bn = out + (size_t)ROWS * (NHD + NPC); // [T,B,256]
    float* out_y_rn = out_y_bn + (size_t)ROWS * NH;     // [T,B,256]
    float* out_s    = out_y_rn + (size_t)ROWS * NH;     // [T,B,256]

    _Float16* w16 = (_Float16*)d_ws;                    // 268*256*2 bytes

    // Phase 1: sequential SNU scan (fused init embeds + recurrence).
    snu_scan_kernel<<<dim3(BB), dim3(NH), 0, stream>>>(
        x, ic, iw, lv, bv, sew, seb, cew, ceb,
        out_y_bn, out_y_rn, out_s);

    // Phase 2: pack readout weights to f16 (independent of phase 1).
    pack_w16_kernel<<<dim3((NW * NH + 255) / 256), dim3(256), 0, stream>>>(
        pc_w, hd_w, w16);

    // Phase 3: batched WMMA readout GEMM over all T*B spike rows.
    //   grid.x = 17 N-tiles (fast-varying -> shared y slice in L2),
    //   grid.y = 8000 M-tiles / (8 waves * MPW) = 125.
    spike_readout_gemm<<<dim3(17, ROWS / 16 / 8 / MPW), dim3(256), 0, stream>>>(
        out_y_bn, w16, pc_b, hd_b, out_pc, out_hd);
}
